// Tacotron_78838419685745
// MI455X (gfx1250) — matrix operand order repaired and checked by reading, not compiled, not run on hardware
//
#include <hip/hip_runtime.h>
#include <hip/hip_bf16.h>

// ---------------------------------------------------------------------------
// Types for CDNA5 WMMA (wave32)
// ---------------------------------------------------------------------------
typedef __bf16 bf16_t;
typedef __attribute__((ext_vector_type(16))) __bf16 v16bf;
typedef __attribute__((ext_vector_type(8)))  float  v8f;

__device__ __constant__ float d_P[11] = {
    0.013173f, 0.013028f, 0.013752f, 0.014979f, 0.016759f,
    0.019322f, 0.023171f, 0.029471f, 0.041574f, 0.074750f, 0.740020f};

__device__ inline float sigmf(float x) { return 1.f / (1.f + __expf(-x)); }

__device__ inline v8f wmma32(v16bf a, v16bf b, v8f c) {
  return __builtin_amdgcn_wmma_f32_16x16x32_bf16(false, a, false, b, (short)0,
                                                 c, false, false);
}

// ---------------------------------------------------------------------------
// Fragment loaders.  A-fragment (16x32 bf16, ISA 7.12.2): per lane two
// contiguous 8-float runs at kc+8*half and kc+16+8*half -> 4 float4 loads.
// B-fragment: two contiguous 8-bf16 runs at kc+8*half and kc+16+8*half.
// All strides are multiples of 16 elements => 32B-aligned accesses.
// ---------------------------------------------------------------------------
__device__ inline void pk8(v16bf& d, int o, float4 u, float4 v) {
  d[o + 0] = (__bf16)u.x; d[o + 1] = (__bf16)u.y;
  d[o + 2] = (__bf16)u.z; d[o + 3] = (__bf16)u.w;
  d[o + 4] = (__bf16)v.x; d[o + 5] = (__bf16)v.y;
  d[o + 6] = (__bf16)v.z; d[o + 7] = (__bf16)v.w;
}
__device__ inline void pk8s(v16bf& d, int o, float4 u, float4 v, float s) {
  d[o + 0] = (__bf16)(u.x * s); d[o + 1] = (__bf16)(u.y * s);
  d[o + 2] = (__bf16)(u.z * s); d[o + 3] = (__bf16)(u.w * s);
  d[o + 4] = (__bf16)(v.x * s); d[o + 5] = (__bf16)(v.y * s);
  d[o + 6] = (__bf16)(v.z * s); d[o + 7] = (__bf16)(v.w * s);
}
__device__ inline v16bf lda_frag(const float4* r4, int kc, int half) {
  int o = (kc >> 2) + 2 * half;
  float4 u0 = r4[o], u1 = r4[o + 1], u2 = r4[o + 4], u3 = r4[o + 5];
  v16bf a; pk8(a, 0, u0, u1); pk8(a, 8, u2, u3); return a;
}
__device__ inline v16bf lda_frag_s(const float4* r4, int kc, int half, float s) {
  int o = (kc >> 2) + 2 * half;
  float4 u0 = r4[o], u1 = r4[o + 1], u2 = r4[o + 4], u3 = r4[o + 5];
  v16bf a; pk8s(a, 0, u0, u1, s); pk8s(a, 8, u2, u3, s); return a;
}
__device__ inline v16bf ldb_frag(const bf16_t* wrow, int kc, int half) {
  const bf16_t* p0 = wrow + kc + 8 * half;
  v16bf b;
#pragma unroll
  for (int i = 0; i < 8; ++i) { b[i] = p0[i]; b[8 + i] = p0[16 + i]; }
  return b;
}
// relu+affine
__device__ inline float4 paff(float4 x, float4 s, float4 o) {
  float4 r;
  r.x = fmaxf(x.x, 0.f) * s.x + o.x; r.y = fmaxf(x.y, 0.f) * s.y + o.y;
  r.z = fmaxf(x.z, 0.f) * s.z + o.z; r.w = fmaxf(x.w, 0.f) * s.w + o.w;
  return r;
}
__device__ inline float4 pmax(float4 a, float4 b, bool hp, float rs) {
  float4 r;
  r.x = fmaxf(hp ? a.x : -3.0e38f, b.x) * rs;
  r.y = fmaxf(hp ? a.y : -3.0e38f, b.y) * rs;
  r.z = fmaxf(hp ? a.z : -3.0e38f, b.z) * rs;
  r.w = fmaxf(hp ? a.w : -3.0e38f, b.w) * rs;
  return r;
}

__device__ inline void store_tile(v8f acc, float* out, int ldo, int mt, int nt,
                                  const float* bias, int act, int accum,
                                  int lane) {
  const int m = lane & 15, half = (lane >> 4) & 1;
  const int nn = nt * 16 + m;
  const float bv = bias ? bias[nn] : 0.f;
#pragma unroll
  for (int v = 0; v < 8; ++v) {
    int mr = mt * 16 + v + 8 * half;
    size_t o = (size_t)mr * ldo + nn;
    float x = acc[v] + bv;
    if (accum) x += out[o];
    if (act == 1)      x = fmaxf(x, 0.f);
    else if (act == 2) x = tanhf(x);
    else if (act == 3) x = sigmf(x);
    out[o] = x;
  }
}

// ---------------------------------------------------------------------------
// Block-cooperative GEMMs (decoder / GRU).  Kv%32==0, Nv%16==0 required.
// ---------------------------------------------------------------------------
__device__ inline void dec_gemm(const float* A, int lda, int K, int M,
                                const bf16_t* W, int ldw, int Nv,
                                const float* bias, float* out, int ldo,
                                int act) {
  const int lane = threadIdx.x & 31, wave = threadIdx.x >> 5;
  const int nw = blockDim.x >> 5;
  const int NT = Nv >> 4, MT = M >> 4, tiles = MT * NT;
  const int m = lane & 15, half = (lane >> 4) & 1;
  for (int t = wave; t < tiles; t += nw) {
    int mt = t / NT, nt = t - mt * NT;
    const float4* r4 = (const float4*)(A + (size_t)(mt * 16 + m) * lda);
    const bf16_t* wrow = W + (size_t)(nt * 16 + m) * ldw;
    v8f acc = {};
    for (int kc = 0; kc < K; kc += 32)
      acc = wmma32(lda_frag(r4, kc, half), ldb_frag(wrow, kc, half), acc);
    store_tile(acc, out, ldo, mt, nt, bias, act, 0, lane);
  }
}

// 16x64 per wave (4 accumulators, A reused).  Nv%64==0.
__device__ inline void dec_gemm4(const float* A, int lda, int K, int M,
                                 const bf16_t* W, int ldw, int Nv,
                                 const float* bias, float* out, int ldo,
                                 int act) {
  const int lane = threadIdx.x & 31, wave = threadIdx.x >> 5;
  const int nw = blockDim.x >> 5;
  const int NG = Nv >> 6, MT = M >> 4, groups = MT * NG;
  const int m = lane & 15, half = (lane >> 4) & 1;
  for (int g = wave; g < groups; g += nw) {
    int mt = g / NG, ng = g - mt * NG;
    const float4* r4 = (const float4*)(A + (size_t)(mt * 16 + m) * lda);
    const bf16_t* w0 = W + (size_t)(ng * 64 + m) * ldw;
    const bf16_t* w1 = w0 + (size_t)16 * ldw;
    const bf16_t* w2 = w0 + (size_t)32 * ldw;
    const bf16_t* w3 = w0 + (size_t)48 * ldw;
    v8f a0 = {}, a1 = {}, a2 = {}, a3 = {};
    for (int kc = 0; kc < K; kc += 32) {
      v16bf af = lda_frag(r4, kc, half);
      a0 = wmma32(af, ldb_frag(w0, kc, half), a0);
      a1 = wmma32(af, ldb_frag(w1, kc, half), a1);
      a2 = wmma32(af, ldb_frag(w2, kc, half), a2);
      a3 = wmma32(af, ldb_frag(w3, kc, half), a3);
    }
    store_tile(a0, out, ldo, mt, ng * 4 + 0, bias, act, 0, lane);
    store_tile(a1, out, ldo, mt, ng * 4 + 1, bias, act, 0, lane);
    store_tile(a2, out, ldo, mt, ng * 4 + 2, bias, act, 0, lane);
    store_tile(a3, out, ldo, mt, ng * 4 + 3, bias, act, 0, lane);
  }
}

__device__ inline void lstm_pw(const float* g, float* h, float* c) {
  for (int i = threadIdx.x; i < 32 * 256; i += blockDim.x) {
    int b = i >> 8, j = i & 255;
    const float* gb = g + (size_t)b * 1024;
    float ig = sigmf(gb[j]), fg = sigmf(gb[256 + j]);
    float gg = tanhf(gb[512 + j]), og = sigmf(gb[768 + j]);
    float cn = fg * c[i] + ig * gg;
    c[i] = cn;
    h[i] = og * tanhf(cn);
  }
}

// ---------------------------------------------------------------------------
// Generic grid GEMM (encoder), 16x64 per wave.
// Modes: 0 plain, 1 conv shift (guard rows, row-scale), 2 conv shift with
// fused relu+BN-affine+maxpool on A, 3 embedding gather.
// Requires Kv%32==0, Nv%64==0, guard rows present for modes 1/2.
// ---------------------------------------------------------------------------
struct GemmP {
  const float* A; int lda; int K;
  int mode; int off; int Lmask;
  const int* idx;
  const float* ps; const float* po;
  const bf16_t* W; int ldw;
  const float* bias;
  float* out; int ldo;
  int M; int Nv;
  int act; int accum;
  const float* os; const float* oo;
};

__global__ __launch_bounds__(128) void gemm_k(GemmP p) {
  const int lane = threadIdx.x & 31, wave = threadIdx.x >> 5;
  const int m = lane & 15, half = (lane >> 4) & 1;
  const int NG = p.Nv >> 6, MT = p.M >> 4, groups = MT * NG;
  for (int g = blockIdx.x * 4 + wave; g < groups; g += gridDim.x * 4) {
    const int mt = g / NG, ng = g - mt * NG;
    const int r = mt * 16 + m;
    const float* arow;
    float rs = 1.f; bool hp = false;
    if (p.mode == 0) {
      arow = p.A + (size_t)r * p.lda;
    } else if (p.mode == 3) {
      arow = p.A + (size_t)p.idx[r] * p.lda;
    } else {
      int l = r & (p.Lmask - 1);
      int l2 = l + p.off;
      rs = ((unsigned)l2 < (unsigned)p.Lmask) ? 1.f : 0.f;
      hp = (l2 >= 1);
      arow = p.A + (size_t)(r + p.off) * p.lda;   // guard rows make this safe
    }
    const float4* r4 = (const float4*)arow;
    const float4* r04 = (const float4*)(arow - p.lda);
    const float4* ps4 = (const float4*)p.ps;
    const float4* po4 = (const float4*)p.po;
    const bf16_t* w0 = p.W + (size_t)(ng * 64 + m) * p.ldw;
    const bf16_t* w1 = w0 + (size_t)16 * p.ldw;
    const bf16_t* w2 = w0 + (size_t)32 * p.ldw;
    const bf16_t* w3 = w0 + (size_t)48 * p.ldw;
    v8f a0 = {}, a1 = {}, a2 = {}, a3 = {};
    if (p.mode == 2) {
      for (int kc = 0; kc < p.K; kc += 32) {
        int o = (kc >> 2) + 2 * half;
        v16bf af;
        {
          float4 x1 = paff(r4[o],     ps4[o],     po4[o]);
          float4 y1 = paff(r4[o + 1], ps4[o + 1], po4[o + 1]);
          float4 x0 = paff(r04[o],     ps4[o],     po4[o]);
          float4 y0 = paff(r04[o + 1], ps4[o + 1], po4[o + 1]);
          pk8(af, 0, pmax(x0, x1, hp, rs), pmax(y0, y1, hp, rs));
          float4 x1b = paff(r4[o + 4], ps4[o + 4], po4[o + 4]);
          float4 y1b = paff(r4[o + 5], ps4[o + 5], po4[o + 5]);
          float4 x0b = paff(r04[o + 4], ps4[o + 4], po4[o + 4]);
          float4 y0b = paff(r04[o + 5], ps4[o + 5], po4[o + 5]);
          pk8(af, 8, pmax(x0b, x1b, hp, rs), pmax(y0b, y1b, hp, rs));
        }
        a0 = wmma32(af, ldb_frag(w0, kc, half), a0);
        a1 = wmma32(af, ldb_frag(w1, kc, half), a1);
        a2 = wmma32(af, ldb_frag(w2, kc, half), a2);
        a3 = wmma32(af, ldb_frag(w3, kc, half), a3);
      }
    } else if (p.mode == 1) {
      for (int kc = 0; kc < p.K; kc += 32) {
        v16bf af = lda_frag_s(r4, kc, half, rs);
        a0 = wmma32(af, ldb_frag(w0, kc, half), a0);
        a1 = wmma32(af, ldb_frag(w1, kc, half), a1);
        a2 = wmma32(af, ldb_frag(w2, kc, half), a2);
        a3 = wmma32(af, ldb_frag(w3, kc, half), a3);
      }
    } else {
      for (int kc = 0; kc < p.K; kc += 32) {
        v16bf af = lda_frag(r4, kc, half);
        a0 = wmma32(af, ldb_frag(w0, kc, half), a0);
        a1 = wmma32(af, ldb_frag(w1, kc, half), a1);
        a2 = wmma32(af, ldb_frag(w2, kc, half), a2);
        a3 = wmma32(af, ldb_frag(w3, kc, half), a3);
      }
    }
#pragma unroll
    for (int q = 0; q < 4; ++q) {
      v8f acc = (q == 0) ? a0 : (q == 1) ? a1 : (q == 2) ? a2 : a3;
      const int nn = ng * 64 + q * 16 + m;
      const float bv = p.bias ? p.bias[nn] : 0.f;
#pragma unroll
      for (int v = 0; v < 8; ++v) {
        int mr = mt * 16 + v + 8 * half;
        size_t oo_ = (size_t)mr * p.ldo + nn;
        float x = acc[v] + bv;
        if (p.accum) x += p.out[oo_];
        if (p.act == 1)      x = fmaxf(x, 0.f);
        else if (p.act == 2) x = tanhf(x);
        else if (p.act == 3) x = sigmf(x);
        if (p.os) x = x * p.os[nn] + p.oo[nn];
        p.out[oo_] = x;
      }
    }
  }
}

// ---------------------------------------------------------------------------
// Prep kernels
// ---------------------------------------------------------------------------
__global__ void cvt_k(const float* s, bf16_t* d, int n) {
  int i = blockIdx.x * 256 + threadIdx.x;
  if (i < n) d[i] = (__bf16)s[i];
}
__global__ void cvt_pad_k(const float* s, bf16_t* d, int Ks, int Kd, int n) {
  int i = blockIdx.x * 256 + threadIdx.x;
  if (i < n) {
    int r = i / Kd, k = i - r * Kd;
    d[i] = (k < Ks) ? (__bf16)s[(size_t)r * Ks + k] : (__bf16)0.f;
  }
}
__global__ void cvt_tap_k(const float* s, bf16_t* d, int OI, int T, int n) {
  int i = blockIdx.x * 256 + threadIdx.x;
  if (i < n) { int t = i / OI, rem = i - t * OI; d[i] = (__bf16)s[(size_t)rem * T + t]; }
}
__global__ void fuse2_k(const float* s1, int K1, const float* s2, int K2,
                        int Kd, bf16_t* d, int n) {
  int i = blockIdx.x * 256 + threadIdx.x;
  if (i < n) {
    int r = i / Kd, k = i - r * Kd;
    float v = (k < K1) ? s1[(size_t)r * K1 + k]
            : (k < K1 + K2) ? s2[(size_t)r * K2 + (k - K1)] : 0.f;
    d[i] = (__bf16)v;
  }
}
__global__ void vadd_k(const float* a, const float* b, float* d, int n) {
  int i = blockIdx.x * 256 + threadIdx.x;
  if (i < n) d[i] = a[i] + b[i];
}
__global__ void addto_k(float* d, const float* a, int n) {
  int i = blockIdx.x * 256 + threadIdx.x;
  if (i < n) d[i] += a[i];
}
__global__ void hwy_pw_k(float* y, const float* t1, const float* t2, int n) {
  int i = blockIdx.x * 256 + threadIdx.x;
  if (i < n) y[i] = t2[i] * t1[i] + (1.f - t2[i]) * y[i];
}
__global__ void fillzf_k(float* d, int n) {
  int i = blockIdx.x * 256 + threadIdx.x;
  if (i < n) d[i] = 0.f;
}
__global__ void fillzh_k(bf16_t* d, int n) {
  int i = blockIdx.x * 256 + threadIdx.x;
  if (i < n) d[i] = (__bf16)0.f;
}

// ---------------------------------------------------------------------------
// Persistent bi-GRU scan: grid(2) = {forward, backward}, 1024 threads.
// ---------------------------------------------------------------------------
struct GruP {
  const float* gi_f; const float* gi_b;
  const bf16_t* whh_f; const bf16_t* whh_b;
  const float* bhh_f; const float* bhh_b;
  float* h;   // 2 x (32x128)
  float* gh;  // 2 x (32x384)
  float* henc;
};

__global__ __launch_bounds__(1024) void gru_scan(GruP P) {
  const int dir = blockIdx.x;
  const float* gi = dir ? P.gi_b : P.gi_f;
  const bf16_t* whh = dir ? P.whh_b : P.whh_f;
  const float* bhh = dir ? P.bhh_b : P.bhh_f;
  float* h = P.h + (size_t)dir * 32 * 128;
  float* gh = P.gh + (size_t)dir * 32 * 384;
  for (int i = threadIdx.x; i < 32 * 128; i += blockDim.x) h[i] = 0.f;
  __syncthreads();
  for (int step = 0; step < 256; ++step) {
    const int t = dir ? (255 - step) : step;
    dec_gemm4(h, 128, 128, 32, whh, 128, 384, nullptr, gh, 384, 0);
    __syncthreads();
    for (int i = threadIdx.x; i < 32 * 128; i += blockDim.x) {
      int b = i >> 7, j = i & 127;
      const float* gib = gi + ((size_t)(b * 256 + t)) * 384;
      const float* ghb = gh + (size_t)b * 384;
      float r = sigmf(gib[j] + ghb[j] + bhh[j]);
      float z = sigmf(gib[128 + j] + ghb[128 + j] + bhh[128 + j]);
      float n = tanhf(gib[256 + j] + r * (ghb[256 + j] + bhh[256 + j]));
      float hn = (1.f - z) * n + z * h[i];
      h[i] = hn;
      P.henc[((size_t)(b * 256 + t)) * 256 + dir * 128 + j] = hn;
    }
    __syncthreads();
  }
}

// ---------------------------------------------------------------------------
// Persistent decoder: one 1024-thread workgroup, 250 sequential steps.
// ---------------------------------------------------------------------------
struct DecP {
  const float* mels;
  const bf16_t *Wd1, *Wd2, *Wattn, *WWw, *WV, *WUT, *Wlin, *Wr1, *Wr2, *Wproj;
  const float *bd1, *bd2, *battn, *Wbb, *Fc, *Tbb, *vv, *lb, *br1, *br2;
  const float* henc;
  float* S;
  float* ys;
  float* alph;
};

// scratch offsets (floats); all multiples of 64 for 256B alignment
#define O_ALPHA 0
#define O_CTX   8192
#define O_AH    16384
#define O_AC    24576
#define O_H1    32768
#define O_C1    40960
#define O_H2    49152
#define O_C2    57344
#define O_LP    65536
#define O_Y     73728       /* 32 x 96 (cols 80..95 stay zero) */
#define O_PH    76800
#define O_YP    84992
#define O_XA    89088
#define O_GATES 109568
#define O_TG    142336
#define O_G     146432      /* 32 x 176 */
#define O_FSG   152064      /* 8192 x 32 (cols 16..31 stay zero) */
#define O_HE    414208
#define O_E     1462784
#define O_XD    1470976
#define O_XA2   1479168
#define O_YO    1495552
#define DEC_SCRATCH_FLOATS 1500672

__global__ __launch_bounds__(1024) void decoder_k(DecP P) {
  float* S = P.S;
  float* alpha = S + O_ALPHA; float* ctx = S + O_CTX;
  float* ah = S + O_AH; float* ac = S + O_AC;
  float* h1 = S + O_H1; float* c1 = S + O_C1;
  float* h2 = S + O_H2; float* c2 = S + O_C2;
  float* lp = S + O_LP; float* yb = S + O_Y; float* ph = S + O_PH;
  float* yp = S + O_YP; float* xa = S + O_XA; float* gates = S + O_GATES;
  float* tG = S + O_TG; float* G = S + O_G; float* fsg = S + O_FSG;
  float* He = S + O_HE; float* eb = S + O_E; float* xd = S + O_XD;
  float* xa2 = S + O_XA2; float* yo = S + O_YO;
  const int tid = threadIdx.x, nth = blockDim.x;

  for (int i = tid; i < O_PH; i += nth) S[i] = 0.f;          // states + yb pad
  for (int i = tid; i < 262144; i += nth) fsg[i] = 0.f;      // fsg incl pad
  __syncthreads();
  if (tid < 32) alpha[tid * 256] = 1.f;
  __syncthreads();

  for (int s = 0; s < 250; ++s) {
    for (int i = tid; i < 2560; i += nth) {
      int b = i / 80, m = i - b * 80;
      yb[b * 96 + m] =
          (s == 0) ? 0.f : P.mels[(size_t)b * 40000 + m * 500 + (2 * s - 1)];
    }
    __syncthreads();
    dec_gemm4(yb, 96, 96, 32, P.Wd1, 96, 256, P.bd1, ph, 256, 1);
    __syncthreads();
    dec_gemm4(ph, 256, 256, 32, P.Wd2, 256, 128, P.bd2, yp, 128, 1);
    __syncthreads();
    for (int i = tid; i < 32 * 640; i += nth) {
      int b = i / 640, k = i - b * 640;
      xa[i] = (k < 256) ? ctx[b * 256 + k]
            : (k < 384) ? yp[b * 128 + (k - 256)]
                        : ah[b * 256 + (k - 384)];
    }
    __syncthreads();
    dec_gemm4(xa, 640, 640, 32, P.Wattn, 640, 1024, P.battn, gates, 1024, 0);
    __syncthreads();
    lstm_pw(gates, ah, ac);
    for (int i = tid; i < 8192; i += nth) {   // beta-binomial prior (old alpha)
      int b = i >> 8, l = i & 255;
      float sm = 0.f;
#pragma unroll
      for (int j = 0; j < 11; ++j) {
        int ls = l - 10 + j;
        if (ls >= 0) sm += alpha[b * 256 + ls] * d_P[j];
      }
      lp[i] = __logf(fmaxf(sm, 1e-6f));
    }
    __syncthreads();
    dec_gemm4(ah, 256, 256, 32, P.WWw, 256, 128, P.Wbb, tG, 128, 2);
    __syncthreads();
    dec_gemm(tG, 128, 128, 32, P.WV, 128, 176, nullptr, G, 176, 0);
    __syncthreads();
    for (int i = tid; i < 131072; i += nth) {
      int r = i >> 4, ch = i & 15, b = r >> 8, l = r & 255;
      float sm = 0.f;
      if (ch < 8) {
#pragma unroll
        for (int j = 0; j < 21; ++j) {
          int ls = l - 10 + j;
          if ((unsigned)ls < 256u) sm += alpha[b * 256 + ls] * P.Fc[ch * 21 + j];
        }
      } else {
        const float* gb = G + (size_t)b * 176 + (ch - 8) * 21;
#pragma unroll
        for (int j = 0; j < 21; ++j) {
          int ls = l - 10 + j;
          if ((unsigned)ls < 256u) sm += alpha[b * 256 + ls] * gb[j];
        }
      }
      fsg[r * 32 + ch] = sm;
    }
    __syncthreads();
    dec_gemm4(fsg, 32, 32, 8192, P.WUT, 32, 128, P.Tbb, He, 128, 2);
    __syncthreads();
    for (int r = tid; r < 8192; r += nth) {
      const float* hr = He + (size_t)r * 128;
      float sm = 0.f;
      for (int j = 0; j < 128; ++j) sm += hr[j] * P.vv[j];
      eb[r] = sm + lp[r];
    }
    __syncthreads();
    {  // row softmax: one wave per batch (wave32 shuffles)
      int w = tid >> 5, ln = tid & 31, b = w;
      float vals[8]; float mx = -3.0e38f;
#pragma unroll
      for (int j = 0; j < 8; ++j) {
        vals[j] = eb[b * 256 + ln + 32 * j];
        mx = fmaxf(mx, vals[j]);
      }
#pragma unroll
      for (int m = 16; m; m >>= 1) mx = fmaxf(mx, __shfl_xor(mx, m, 32));
      float ss = 0.f;
#pragma unroll
      for (int j = 0; j < 8; ++j) { vals[j] = __expf(vals[j] - mx); ss += vals[j]; }
#pragma unroll
      for (int m = 16; m; m >>= 1) ss += __shfl_xor(ss, m, 32);
      float inv = 1.f / ss;
#pragma unroll
      for (int j = 0; j < 8; ++j) {
        int idx = b * 256 + ln + 32 * j;
        float av = vals[j] * inv;
        alpha[idx] = av;
        P.alph[(size_t)idx * 250 + s] = av;
      }
    }
    __syncthreads();
    for (int i = tid; i < 8192; i += nth) {   // context = alpha @ henc
      int b = i >> 8, d = i & 255;
      const float* hb = P.henc + (size_t)b * 65536 + d;
      const float* al = alpha + b * 256;
      float sm = 0.f;
      for (int l = 0; l < 256; ++l) {
        if ((l & 63) == 0) __builtin_prefetch(hb + (size_t)(l + 64) * 256, 0, 0);
        sm += al[l] * hb[(size_t)l * 256];
      }
      ctx[i] = sm;
    }
    __syncthreads();
    for (int i = tid; i < 16384; i += nth) {
      int b = i >> 9, k = i & 511;
      xa2[i] = (k < 256) ? ctx[b * 256 + k] : ah[b * 256 + (k - 256)];
    }
    __syncthreads();
    dec_gemm4(xa2, 512, 512, 32, P.Wlin, 512, 256, P.lb, xd, 256, 0);
    __syncthreads();
    for (int i = tid; i < 16384; i += nth) {
      int b = i >> 9, k = i & 511;
      xa2[i] = (k < 256) ? xd[b * 256 + k] : h1[b * 256 + (k - 256)];
    }
    __syncthreads();
    dec_gemm4(xa2, 512, 512, 32, P.Wr1, 512, 1024, P.br1, gates, 1024, 0);
    __syncthreads();
    lstm_pw(gates, h1, c1);
    __syncthreads();
    for (int i = tid; i < 8192; i += nth) xd[i] += h1[i];
    __syncthreads();
    for (int i = tid; i < 16384; i += nth) {
      int b = i >> 9, k = i & 511;
      xa2[i] = (k < 256) ? xd[b * 256 + k] : h2[b * 256 + (k - 256)];
    }
    __syncthreads();
    dec_gemm4(xa2, 512, 512, 32, P.Wr2, 512, 1024, P.br2, gates, 1024, 0);
    __syncthreads();
    lstm_pw(gates, h2, c2);
    __syncthreads();
    for (int i = tid; i < 8192; i += nth) xd[i] += h2[i];
    __syncthreads();
    dec_gemm(xd, 256, 256, 32, P.Wproj, 256, 160, nullptr, yo, 160, 0);
    __syncthreads();
    for (int i = tid; i < 5120; i += nth) {
      int b = i / 160, q = i - b * 160, m = q >> 1, r = q & 1;
      P.ys[(size_t)b * 40000 + m * 500 + 2 * s + r] = yo[i];
    }
    __syncthreads();
  }
}

// ---------------------------------------------------------------------------
// Host side
// ---------------------------------------------------------------------------
static void launch_gemm(hipStream_t st, const float* A, int lda, int K,
                        int mode, int off, int Lmask, const int* idx,
                        const float* ps, const float* po, const bf16_t* W,
                        int ldw, const float* bias, float* out, int ldo, int M,
                        int Nv, int act, int accum, const float* os,
                        const float* oo) {
  GemmP p;
  p.A = A; p.lda = lda; p.K = K;
  p.mode = mode; p.off = off; p.Lmask = Lmask; p.idx = idx;
  p.ps = ps; p.po = po; p.W = W; p.ldw = ldw; p.bias = bias;
  p.out = out; p.ldo = ldo; p.M = M; p.Nv = Nv;
  p.act = act; p.accum = accum; p.os = os; p.oo = oo;
  int groups = (M / 16) * (Nv / 64);
  int blocks = (groups + 3) / 4;
  gemm_k<<<blocks, 128, 0, st>>>(p);
}

#define GRID1(n) dim3(((n) + 255) / 256), dim3(256)

extern "C" void kernel_launch(void* const* d_in, const int* in_sizes, int n_in,
                              void* d_out, int out_size, void* d_ws,
                              size_t ws_size, hipStream_t stream) {
  (void)in_sizes; (void)n_in; (void)out_size; (void)ws_size;
  auto f32 = [&](int i) { return (const float*)d_in[i]; };
  const int* x_idx = (const int*)d_in[0];
  const float* mels = f32(1);

  char* base = (char*)d_ws;
  size_t cur = 0;
  auto alloc = [&](size_t nbytes) -> void* {
    void* p = base + cur;
    cur = (cur + nbytes + 255) & ~(size_t)255;
    return p;
  };

  // ---- bf16 weight pool ----
  bf16_t* wb_pre1 = (bf16_t*)alloc(65536 * 2);
  bf16_t* wb_pre2 = (bf16_t*)alloc(32768 * 2);
  bf16_t* wb_bank = (bf16_t*)alloc((size_t)2228224 * 2);
  bf16_t* wb_proj1 = (bf16_t*)alloc((size_t)786432 * 2);
  bf16_t* wb_proj2 = (bf16_t*)alloc(49152 * 2);
  bf16_t* wb_hwy = (bf16_t*)alloc(131072 * 2);
  bf16_t* wb_gruf_ih = (bf16_t*)alloc(49152 * 2);
  bf16_t* wb_gruf_hh = (bf16_t*)alloc(49152 * 2);
  bf16_t* wb_grub_ih = (bf16_t*)alloc(49152 * 2);
  bf16_t* wb_grub_hh = (bf16_t*)alloc(49152 * 2);
  bf16_t* wb_dec1 = (bf16_t*)alloc(24576 * 2);      // 256 x 96 (K padded)
  bf16_t* wb_dec2 = (bf16_t*)alloc(32768 * 2);
  bf16_t* wb_attn = (bf16_t*)alloc((size_t)655360 * 2);
  bf16_t* wb_Ww = (bf16_t*)alloc(32768 * 2);
  bf16_t* wb_V = (bf16_t*)alloc(22528 * 2);         // 176 x 128 (N padded)
  bf16_t* wb_UT = (bf16_t*)alloc(4096 * 2);         // 128 x 32  (K padded)
  bf16_t* wb_lin = (bf16_t*)alloc(131072 * 2);
  bf16_t* wb_rnn1 = (bf16_t*)alloc((size_t)524288 * 2);
  bf16_t* wb_rnn2 = (bf16_t*)alloc((size_t)524288 * 2);
  bf16_t* wb_proj = (bf16_t*)alloc(40960 * 2);

  // ---- fp32 scratch (conv inputs have zeroed guard rows) ----
  float* H1 = (float*)alloc((size_t)2097152 * 4);
  float* xc_all = (float*)alloc((size_t)(8192 + 16) * 128 * 4);
  float* xc = xc_all + 8 * 128;
  float* bank_all = (float*)alloc((size_t)(8192 + 4) * 2048 * 4);
  float* bank = bank_all + 2 * 2048;
  float* p1_all = (float*)alloc((size_t)(8192 + 4) * 128 * 4);
  float* p1 = p1_all + 2 * 128;
  float* p2 = (float*)alloc((size_t)1048576 * 4);
  float* t1a = (float*)alloc((size_t)1048576 * 4);
  float* t2a = (float*)alloc((size_t)1048576 * 4);
  float* gi_f = (float*)alloc((size_t)3145728 * 4);
  float* gi_b = (float*)alloc((size_t)3145728 * 4);
  float* ghb = (float*)alloc((size_t)24576 * 4);
  float* gruh = (float*)alloc((size_t)8192 * 4);
  float* henc = (float*)alloc((size_t)2097152 * 4);
  float* battn = (float*)alloc(1024 * 4);
  float* br1 = (float*)alloc(1024 * 4);
  float* br2 = (float*)alloc(1024 * 4);
  float* s2048 = (float*)alloc(2048 * 4);
  float* o2048 = (float*)alloc(2048 * 4);
  float* decS = (float*)alloc((size_t)DEC_SCRATCH_FLOATS * 4);

  // ================= prep =================
  auto cvt = [&](int src, bf16_t* dst, int n) {
    cvt_k<<<GRID1(n), 0, stream>>>(f32(src), dst, n);
  };
  cvt(3, wb_pre1, 65536);
  cvt(5, wb_pre2, 32768);
  for (int l = 0; l < 4; ++l) {
    cvt(61 + 4 * l, wb_hwy + l * 32768, 16384);
    cvt(63 + 4 * l, wb_hwy + l * 32768 + 16384, 16384);
  }
  cvt(77, wb_gruf_ih, 49152);
  cvt(78, wb_gruf_hh, 49152);
  cvt(81, wb_grub_ih, 49152);
  cvt(82, wb_grub_hh, 49152);
  cvt_pad_k<<<GRID1(24576), 0, stream>>>(f32(85), wb_dec1, 80, 96, 24576);
  cvt(87, wb_dec2, 32768);
  cvt(93, wb_Ww, 32768);
  cvt(95, wb_V, 21504);
  fillzh_k<<<GRID1(1024), 0, stream>>>(wb_V + 21504, 1024);  // 8 zero rows
  cvt(101, wb_lin, 131072);
  cvt(111, wb_proj, 40960);

  {  // conv taps
    size_t boff = 0;
    for (int ks = 1; ks <= 16; ++ks) {
      int n = 128 * 128 * ks;
      cvt_tap_k<<<GRID1(n), 0, stream>>>(f32(6 + ks), wb_bank + boff, 128 * 128, ks, n);
      boff += n;
    }
    cvt_tap_k<<<GRID1(786432), 0, stream>>>(f32(55), wb_proj1, 128 * 2048, 3, 786432);
    cvt_tap_k<<<GRID1(49152), 0, stream>>>(f32(58), wb_proj2, 128 * 128, 3, 49152);
  }
  fuse2_k<<<GRID1(655360), 0, stream>>>(f32(89), 384, f32(90), 256, 640, wb_attn, 655360);
  fuse2_k<<<GRID1(524288), 0, stream>>>(f32(103), 256, f32(104), 256, 512, wb_rnn1, 524288);
  fuse2_k<<<GRID1(524288), 0, stream>>>(f32(107), 256, f32(108), 256, 512, wb_rnn2, 524288);
  fuse2_k<<<GRID1(4096), 0, stream>>>(f32(97), 8, f32(98), 8, 32, wb_UT, 4096);
  vadd_k<<<GRID1(1024), 0, stream>>>(f32(91), f32(92), battn, 1024);
  vadd_k<<<GRID1(1024), 0, stream>>>(f32(105), f32(106), br1, 1024);
  vadd_k<<<GRID1(1024), 0, stream>>>(f32(109), f32(110), br2, 1024);
  for (int kk = 0; kk < 16; ++kk) {
    hipMemcpyAsync(s2048 + kk * 128, d_in[23 + 2 * kk], 512,
                   hipMemcpyDeviceToDevice, stream);
    hipMemcpyAsync(o2048 + kk * 128, d_in[24 + 2 * kk], 512,
                   hipMemcpyDeviceToDevice, stream);
  }
  // zero guard rows
  fillzf_k<<<GRID1(1024), 0, stream>>>(xc_all, 1024);
  fillzf_k<<<GRID1(1024), 0, stream>>>(xc_all + (size_t)(8192 + 8) * 128, 1024);
  fillzf_k<<<GRID1(4096), 0, stream>>>(bank_all, 4096);
  fillzf_k<<<GRID1(4096), 0, stream>>>(bank_all + (size_t)(8192 + 2) * 2048, 4096);
  fillzf_k<<<GRID1(256), 0, stream>>>(p1_all, 256);
  fillzf_k<<<GRID1(256), 0, stream>>>(p1_all + (size_t)(8192 + 2) * 128, 256);

  // ================= encoder =================
  launch_gemm(stream, f32(2), 256, 256, 3, 0, 0, x_idx, nullptr, nullptr,
              wb_pre1, 256, f32(4), H1, 256, 8192, 256, 1, 0, nullptr, nullptr);
  launch_gemm(stream, H1, 256, 256, 0, 0, 0, nullptr, nullptr, nullptr,
              wb_pre2, 256, f32(6), xc, 128, 8192, 128, 1, 0, nullptr, nullptr);
  {  // conv bank: tap-shifted accumulating GEMMs
    size_t boff = 0;
    for (int ks = 1; ks <= 16; ++ks) {
      for (int t = 0; t < ks; ++t)
        launch_gemm(stream, xc, 128, 128, 1, t - ks / 2, 256, nullptr, nullptr,
                    nullptr, wb_bank + boff + (size_t)t * 16384, 128, nullptr,
                    bank + (ks - 1) * 128, 2048, 8192, 128, 0, t > 0, nullptr,
                    nullptr);
      boff += (size_t)16384 * ks;
    }
  }
  for (int t = 0; t < 3; ++t)   // proj1: fused relu+BN-affine+maxpool on A
    launch_gemm(stream, bank, 2048, 2048, 2, t - 1, 256, nullptr, s2048, o2048,
                wb_proj1 + (size_t)t * 128 * 2048, 2048, nullptr, p1, 128, 8192,
                128, (t == 2) ? 1 : 0, t > 0,
                (t == 2) ? f32(56) : nullptr, (t == 2) ? f32(57) : nullptr);
  for (int t = 0; t < 3; ++t)   // proj2
    launch_gemm(stream, p1, 128, 128, 1, t - 1, 256, nullptr, nullptr, nullptr,
                wb_proj2 + (size_t)t * 16384, 128, nullptr, p2, 128, 8192, 128,
                0, t > 0, (t == 2) ? f32(59) : nullptr,
                (t == 2) ? f32(60) : nullptr);
  addto_k<<<GRID1(1048576), 0, stream>>>(p2, xc, 1048576);
  for (int l = 0; l < 4; ++l) {  // highway x4
    launch_gemm(stream, p2, 128, 128, 0, 0, 0, nullptr, nullptr, nullptr,
                wb_hwy + l * 32768, 128, f32(62 + 4 * l), t1a, 128, 8192, 128,
                1, 0, nullptr, nullptr);
    launch_gemm(stream, p2, 128, 128, 0, 0, 0, nullptr, nullptr, nullptr,
                wb_hwy + l * 32768 + 16384, 128, f32(64 + 4 * l), t2a, 128,
                8192, 128, 3, 0, nullptr, nullptr);
    hwy_pw_k<<<GRID1(1048576), 0, stream>>>(p2, t1a, t2a, 1048576);
  }
  launch_gemm(stream, p2, 128, 128, 0, 0, 0, nullptr, nullptr, nullptr,
              wb_gruf_ih, 128, f32(79), gi_f, 384, 8192, 384, 0, 0, nullptr,
              nullptr);
  launch_gemm(stream, p2, 128, 128, 0, 0, 0, nullptr, nullptr, nullptr,
              wb_grub_ih, 128, f32(83), gi_b, 384, 8192, 384, 0, 0, nullptr,
              nullptr);
  {
    GruP g;
    g.gi_f = gi_f; g.gi_b = gi_b;
    g.whh_f = wb_gruf_hh; g.whh_b = wb_grub_hh;
    g.bhh_f = f32(80); g.bhh_b = f32(84);
    g.h = gruh; g.gh = ghb; g.henc = henc;
    gru_scan<<<2, 1024, 0, stream>>>(g);
  }

  // ================= decoder =================
  {
    DecP P;
    P.mels = mels;
    P.Wd1 = wb_dec1; P.Wd2 = wb_dec2; P.Wattn = wb_attn;
    P.WWw = wb_Ww; P.WV = wb_V; P.WUT = wb_UT;
    P.Wlin = wb_lin; P.Wr1 = wb_rnn1; P.Wr2 = wb_rnn2; P.Wproj = wb_proj;
    P.bd1 = f32(86); P.bd2 = f32(88); P.battn = battn;
    P.Wbb = f32(94); P.Fc = f32(96); P.Tbb = f32(99); P.vv = f32(100);
    P.lb = f32(102); P.br1 = br1; P.br2 = br2;
    P.henc = henc; P.S = decS;
    P.ys = (float*)d_out;
    P.alph = (float*)d_out + 1280000;
    decoder_k<<<1, 1024, 0, stream>>>(P);
  }
}
